// RelativeMultiHeadAttention_23184233464576
// MI455X (gfx1250) — matrix operand order repaired and checked by reading, not compiled, not run on hardware
//
#include <hip/hip_runtime.h>
#include <hip/hip_bf16.h>

#define EMBED  1024
#define NHEAD  8
#define HDIM   128
#define SEQ    1024
#define BATCH  4
#define MAXD   128
#define RADIUS 256
#define RPAD   272   // 257 rounded up to 17*16

typedef __attribute__((ext_vector_type(16))) _Float16     v16h;
typedef __attribute__((ext_vector_type(8)))  float        v8f;
typedef __attribute__((ext_vector_type(4)))  unsigned int v4u;

union FragU { unsigned int u[8]; v4u q[2]; v16h h; };

__device__ __forceinline__ unsigned pkf16(float a, float b) {
  union { _Float16 h[2]; unsigned u; } r;
  r.h[0] = (_Float16)a; r.h[1] = (_Float16)b;
  return r.u;
}

// ---- A fragment (16x32 f16): lane m = L&15, half h = L>>4.
// dword v: K = 2*v + 16*h -> 16 consecutive halves -> two 16-byte loads.
__device__ __forceinline__ v16h wA_f16(const _Float16* src, int ld, int row0, int k0) {
  int lane = threadIdx.x & 31, m = lane & 15, hh = lane >> 4;
  const _Float16* p = src + (long)(row0 + m) * ld + k0 + 16 * hh;
  FragU f;
  f.q[0] = *(const v4u*)(p);
  f.q[1] = *(const v4u*)(p + 8);
  return f.h;
}

// ---- B fragment (32x16 f16) from B^T (rows = N, K contiguous).
// dword v: K = 2*v + 16*h -> 16 consecutive halves -> two 16-byte loads.
__device__ __forceinline__ v16h wBt_f16(const _Float16* bt, int ld, int n0, int k0) {
  int lane = threadIdx.x & 31, nn = lane & 15, hh = lane >> 4;
  const _Float16* p = bt + (long)(n0 + nn) * ld + k0 + 16 * hh;
  FragU f;
  f.q[0] = *(const v4u*)(p);
  f.q[1] = *(const v4u*)(p + 8);
  return f.h;
}
__device__ __forceinline__ v16h wBt_f16_clamp(const _Float16* bt, int ld, int n0, int k0,
                                              int maxrow) {
  int lane = threadIdx.x & 31, nn = lane & 15, hh = lane >> 4;
  int row = n0 + nn; if (row > maxrow) row = maxrow;
  const _Float16* p = bt + (long)row * ld + k0 + 16 * hh;
  FragU f;
  f.q[0] = *(const v4u*)(p);
  f.q[1] = *(const v4u*)(p + 8);
  return f.h;
}

#define WMMA(a, b, c) \
  __builtin_amdgcn_wmma_f32_16x16x32_f16(false, (a), false, (b), (short)0, (c), false, false)

__device__ __forceinline__ void vzero(v8f& v) {
#pragma unroll
  for (int r = 0; r < 8; ++r) v[r] = 0.0f;
}

// ============================================================================
// Kernel 0: one-time fp32 -> f16 conversion (8 elements / thread, b128 I/O).
// ============================================================================
__global__ void __launch_bounds__(256) cvt_kernel(const float* __restrict__ src,
                                                  _Float16* __restrict__ dst, int n) {
  int i = (blockIdx.x * 256 + (int)threadIdx.x) * 8;
  if (i >= n) return;
  if (i + 8 <= n) {
    float4 a = *(const float4*)(src + i);
    float4 b = *(const float4*)(src + i + 4);
    v4u o;
    o.x = pkf16(a.x, a.y); o.y = pkf16(a.z, a.w);
    o.z = pkf16(b.x, b.y); o.w = pkf16(b.z, b.w);
    *(v4u*)(dst + i) = o;
  } else {
    for (int j = i; j < n; ++j) dst[j] = (_Float16)src[j];
  }
}

// ============================================================================
// Kernel 1: Y = Xh @ Wh^T (all f16 operands, fp32 accum).  One wave per 32x64
// tile: 8 WMMA / K-step with A/B reuse; inner loop is pure b128 loads + wmma.
// mode 0/1 -> [bh][s][d]; mode 2 -> [bh][d][s] (transposed V).
// ============================================================================
__global__ void __launch_bounds__(128) proj_kernel(const _Float16* __restrict__ Xh,
                                                   const _Float16* __restrict__ Wh,
                                                   _Float16* __restrict__ dst, int mode) {
  int gw = blockIdx.x * 4 + (threadIdx.x >> 5);
  int mtile = gw >> 4, ntile = gw & 15;
  int m0 = mtile * 32, n0 = ntile * 64;
  v8f c[2][4];
#pragma unroll
  for (int i = 0; i < 2; ++i)
#pragma unroll
    for (int j = 0; j < 4; ++j) vzero(c[i][j]);

  for (int k0 = 0; k0 < EMBED; k0 += 32) {
    v16h a0 = wA_f16(Xh, EMBED, m0, k0);
    v16h a1 = wA_f16(Xh, EMBED, m0 + 16, k0);
#pragma unroll
    for (int j = 0; j < 4; ++j) {
      v16h b = wBt_f16(Wh, EMBED, n0 + 16 * j, k0);
      c[0][j] = WMMA(a0, b, c[0][j]);
      c[1][j] = WMMA(a1, b, c[1][j]);
    }
  }

  int lane = threadIdx.x & 31, nn = lane & 15, hh = lane >> 4;
  if (mode == 2) {  // V: store transposed [bh][d][s]
#pragma unroll
    for (int i = 0; i < 2; ++i)
#pragma unroll
      for (int j = 0; j < 4; ++j)
#pragma unroll
        for (int r = 0; r < 8; ++r) {
          int row = m0 + 16 * i + r + 8 * hh;
          int col = n0 + 16 * j + nn;
          int bb = row >> 10, s = row & 1023;
          int hd = col >> 7, d = col & 127;
          dst[((long)(bb * NHEAD + hd) * HDIM + d) * SEQ + s] = (_Float16)c[i][j][r];
        }
  } else {          // Q/K: store [bh][s][d]
#pragma unroll
    for (int i = 0; i < 2; ++i)
#pragma unroll
      for (int j = 0; j < 4; ++j)
#pragma unroll
        for (int r = 0; r < 8; ++r) {
          int row = m0 + 16 * i + r + 8 * hh;
          int col = n0 + 16 * j + nn;
          int bb = row >> 10, s = row & 1023;
          int hd = col >> 7, d = col & 127;
          dst[((long)(bb * NHEAD + hd) * SEQ + s) * HDIM + d] = (_Float16)c[i][j][r];
        }
  }
}

// ============================================================================
// Kernel 2: Q_rel[bh][q][r] = Qp[bh][q] . rel_emb[h][r], fp32, padded to 272.
// One wave per 16x16 tile; K = 128 (4 WMMA).
// ============================================================================
__global__ void __launch_bounds__(128) qrel_kernel(const _Float16* __restrict__ Qp,
                                                   const _Float16* __restrict__ relh,
                                                   float* __restrict__ Qrel) {
  int gw = blockIdx.x * 4 + (threadIdx.x >> 5);
  if (gw >= 32 * 64 * 17) return;
  int rt = gw % 17; int rest = gw / 17;
  int qb = rest & 63; int bh = rest >> 6;
  int head = bh & 7; int q0 = qb * 16;
  const _Float16* Qb = Qp + (long)bh * SEQ * HDIM;
  const _Float16* Rb = relh + (long)head * 257 * HDIM;
  v8f c; vzero(c);
#pragma unroll
  for (int k0 = 0; k0 < HDIM; k0 += 32) {
    v16h a = wA_f16(Qb, HDIM, q0, k0);
    v16h b = wBt_f16_clamp(Rb, HDIM, rt * 16, k0, 256);
    c = WMMA(a, b, c);
  }
  int lane = threadIdx.x & 31, nn = lane & 15, hh = lane >> 4;
#pragma unroll
  for (int r = 0; r < 8; ++r)
    Qrel[((long)bh * SEQ + q0 + r + 8 * hh) * RPAD + rt * 16 + nn] = c[r];
}

// ============================================================================
// Kernel 3: banded attention with online softmax.
// One wave per (bh, 16-query block). k band = [max(0,q0-256)&~31, min(1024,q0+272)).
// ============================================================================
__global__ void __launch_bounds__(128) attn_kernel(const _Float16* __restrict__ Qp,
                                                   const _Float16* __restrict__ Kp,
                                                   const _Float16* __restrict__ Vt,
                                                   const float* __restrict__ Qrel,
                                                   const unsigned char* __restrict__ kpm,
                                                   _Float16* __restrict__ Ctx) {
  __shared__ __align__(16) _Float16 lds[4][16 * 32];
  int wid = threadIdx.x >> 5;
  int gw = blockIdx.x * 4 + wid;
  int bh = gw >> 6, qb = gw & 63;
  int q0 = qb * 16;
  int b = bh >> 3, head = bh & 7;
  int lane = threadIdx.x & 31, nn = lane & 15, hh = lane >> 4;

  const _Float16* Qb = Qp + (long)bh * SEQ * HDIM;
  const _Float16* Kb = Kp + (long)bh * SEQ * HDIM;
  const _Float16* Vb = Vt + (long)bh * HDIM * SEQ;
  const float*    QR = Qrel + (long)bh * SEQ * RPAD;

  v16h qf[4];
#pragma unroll
  for (int kk = 0; kk < 4; ++kk) qf[kk] = wA_f16(Qb, HDIM, q0, kk * 32);

  v8f O[8];
#pragma unroll
  for (int dt = 0; dt < 8; ++dt) vzero(O[dt]);
  float mrow[8], lrow[8];
#pragma unroll
  for (int r = 0; r < 8; ++r) { mrow[r] = -1.0e30f; lrow[r] = 0.0f; }

  int ks = q0 - RADIUS; if (ks < 0) ks = 0; ks &= ~31;
  int ke = q0 + 16 + RADIUS; if (ke > SEQ) ke = SEQ;
  const float invs = 0.088388347648318447f;  // 1/sqrt(128)
  _Float16* my = lds[wid];

  for (int k = ks; k < ke; k += 32) {
    // energy tiles: cols [k,k+16) and [k+16,k+32)  (k 32-aligned < 1024 => rows < 1024)
    v8f e0, e1; vzero(e0); vzero(e1);
#pragma unroll
    for (int kk = 0; kk < 4; ++kk) {
      v16h b0 = wBt_f16(Kb, HDIM, k, kk * 32);
      e0 = WMMA(qf[kk], b0, e0);
      v16h b1 = wBt_f16(Kb, HDIM, k + 16, kk * 32);
      e1 = WMMA(qf[kk], b1, e1);
    }
    // column-dependent mask bits (shared by all 8 rows of this lane)
    bool kp0 = kpm[b * SEQ + k + nn] != 0;
    bool kp1 = kpm[b * SEQ + k + 16 + nn] != 0;
#pragma unroll
    for (int r = 0; r < 8; ++r) {
      int q = q0 + r + 8 * hh;
      int kc0 = k + nn, kc1 = k + 16 + nn;
      int d0 = kc0 - q, d1 = kc1 - q;
      int i0 = (d0 < -MAXD ? -MAXD : (d0 > MAXD ? MAXD : d0)) + MAXD;
      int i1 = (d1 < -MAXD ? -MAXD : (d1 > MAXD ? MAXD : d1)) + MAXD;
      float v0 = (e0[r] + QR[(long)q * RPAD + i0]) * invs;
      float v1 = (e1[r] + QR[(long)q * RPAD + i1]) * invs;
      bool m0 = (d0 > RADIUS) || (d0 < -RADIUS) || kp0;
      bool m1 = (d1 > RADIUS) || (d1 < -RADIUS) || kp1;
      float le0 = m0 ? -3.0e38f : v0;
      float le1 = m1 ? -3.0e38f : v1;
      // row max over the 16 lanes holding this row's columns
      float rm = fmaxf(le0, le1);
#pragma unroll
      for (int off = 1; off < 16; off <<= 1) rm = fmaxf(rm, __shfl_xor(rm, off, 16));
      float mn = fmaxf(mrow[r], rm);
      float sc = __expf(mrow[r] - mn);
      float p0 = __expf(le0 - mn);
      float p1 = __expf(le1 - mn);
      float rs = p0 + p1;
#pragma unroll
      for (int off = 1; off < 16; off <<= 1) rs += __shfl_xor(rs, off, 16);
      lrow[r] = lrow[r] * sc + rs;
      mrow[r] = mn;
#pragma unroll
      for (int dt = 0; dt < 8; ++dt) O[dt][r] *= sc;
      // stash P in D-layout; re-read below in A-layout (transpose via LDS)
      my[(r + 8 * hh) * 32 + nn]      = (_Float16)p0;
      my[(r + 8 * hh) * 32 + 16 + nn] = (_Float16)p1;
    }
    v16h pf = wA_f16(my, 32, 0, 0);  // same-wave LDS ops are in order
#pragma unroll
    for (int dt = 0; dt < 8; ++dt) {
      v16h vb = wBt_f16(Vb, SEQ, dt * 16, k);  // Vt rows = d, cols = s (contiguous)
      O[dt] = WMMA(pf, vb, O[dt]);
    }
  }

  // normalize + store in reference channel order c = d*NHEAD + head
#pragma unroll
  for (int r = 0; r < 8; ++r) {
    float rl = 1.0f / lrow[r];
    int q = q0 + r + 8 * hh;
    long base = (long)(b * SEQ + q) * EMBED;
#pragma unroll
    for (int dt = 0; dt < 8; ++dt) {
      int d = dt * 16 + nn;
      Ctx[base + d * NHEAD + head] = (_Float16)(O[dt][r] * rl);
    }
  }
}

// ============================================================================
// Kernel 4: out = Ctx @ Woh^T, fp32 output.  One wave per 32x64 tile.
// ============================================================================
__global__ void __launch_bounds__(128) oproj_kernel(const _Float16* __restrict__ Ctx,
                                                    const _Float16* __restrict__ Woh,
                                                    float* __restrict__ out) {
  int gw = blockIdx.x * 4 + (threadIdx.x >> 5);
  int mtile = gw >> 4, ntile = gw & 15;
  int m0 = mtile * 32, n0 = ntile * 64;
  v8f c[2][4];
#pragma unroll
  for (int i = 0; i < 2; ++i)
#pragma unroll
    for (int j = 0; j < 4; ++j) vzero(c[i][j]);

  for (int k0 = 0; k0 < EMBED; k0 += 32) {
    v16h a0 = wA_f16(Ctx, EMBED, m0, k0);
    v16h a1 = wA_f16(Ctx, EMBED, m0 + 16, k0);
#pragma unroll
    for (int j = 0; j < 4; ++j) {
      v16h b = wBt_f16(Woh, EMBED, n0 + 16 * j, k0);
      c[0][j] = WMMA(a0, b, c[0][j]);
      c[1][j] = WMMA(a1, b, c[1][j]);
    }
  }

  int lane = threadIdx.x & 31, nn = lane & 15, hh = lane >> 4;
#pragma unroll
  for (int i = 0; i < 2; ++i)
#pragma unroll
    for (int j = 0; j < 4; ++j)
#pragma unroll
      for (int r = 0; r < 8; ++r) {
        int row = m0 + 16 * i + r + 8 * hh;
        out[(long)row * EMBED + n0 + 16 * j + nn] = c[i][j][r];
      }
}

extern "C" void kernel_launch(void* const* d_in, const int* in_sizes, int n_in,
                              void* d_out, int out_size, void* d_ws, size_t ws_size,
                              hipStream_t stream) {
  (void)in_sizes; (void)n_in; (void)out_size; (void)ws_size;
  const float* Q  = (const float*)d_in[0];
  const float* K  = (const float*)d_in[1];
  const float* V  = (const float*)d_in[2];
  const unsigned char* kpm = (const unsigned char*)d_in[4];  // jnp.bool_ storage
  const float* Wq = (const float*)d_in[5];
  const float* Wk = (const float*)d_in[6];
  const float* Wv = (const float*)d_in[7];
  const float* Wo = (const float*)d_in[8];
  const float* rel = (const float*)d_in[9];

  char* ws = (char*)d_ws;
  _Float16* Qp   = (_Float16*)(ws + 0);          //  8 MB  [32][1024][128]
  _Float16* Kp   = (_Float16*)(ws + 8388608);    //  8 MB  [32][1024][128]
  _Float16* Vt   = (_Float16*)(ws + 16777216);   //  8 MB  [32][128][1024]
  _Float16* Ctx  = (_Float16*)(ws + 25165824);   //  8 MB  [4096][1024]
  float*    Qrl  = (float*)   (ws + 33554432);   // 35.7 MB [32][1024][272]
  _Float16* Xh   = (_Float16*)(ws + 69206016);   //  8 MB  staging (reused Q/K/V)
  _Float16* Wqh  = (_Float16*)(ws + 77594624);   //  2 MB
  _Float16* Wkh  = (_Float16*)(ws + 79691776);   //  2 MB
  _Float16* Wvh  = (_Float16*)(ws + 81788928);   //  2 MB
  _Float16* Woh  = (_Float16*)(ws + 83886080);   //  2 MB
  _Float16* relh = (_Float16*)(ws + 85983232);   //  0.5 MB [8][257][128]

  const int nX = BATCH * SEQ * EMBED;   // 4194304
  const int nW = EMBED * EMBED;         // 1048576
  const int nR = NHEAD * 257 * HDIM;    // 263168
  dim3 blk(128), cblk(256);

  // one-time f16 conversions (weights + rel_emb)
  cvt_kernel<<<nW / 2048, cblk, 0, stream>>>(Wq, Wqh, nW);
  cvt_kernel<<<nW / 2048, cblk, 0, stream>>>(Wk, Wkh, nW);
  cvt_kernel<<<nW / 2048, cblk, 0, stream>>>(Wv, Wvh, nW);
  cvt_kernel<<<nW / 2048, cblk, 0, stream>>>(Wo, Woh, nW);
  cvt_kernel<<<(nR + 2047) / 2048, cblk, 0, stream>>>(rel, relh, nR);

  // projections (X staging reused; launches are in-order on the stream)
  cvt_kernel<<<nX / 2048, cblk, 0, stream>>>(Q, Xh, nX);
  proj_kernel<<<512, blk, 0, stream>>>(Xh, Wqh, Qp, 0);
  cvt_kernel<<<nX / 2048, cblk, 0, stream>>>(K, Xh, nX);
  proj_kernel<<<512, blk, 0, stream>>>(Xh, Wkh, Kp, 1);
  cvt_kernel<<<nX / 2048, cblk, 0, stream>>>(V, Xh, nX);
  proj_kernel<<<512, blk, 0, stream>>>(Xh, Wvh, Vt, 2);

  qrel_kernel<<<8704, blk, 0, stream>>>(Qp, relh, Qrl);
  attn_kernel<<<512, blk, 0, stream>>>(Qp, Kp, Vt, Qrl, kpm, Ctx);
  oproj_kernel<<<512, blk, 0, stream>>>(Ctx, Woh, (float*)d_out);
}
